// RNN_S_21878563406024
// MI455X (gfx1250) — matrix operand order repaired and checked by reading, not compiled, not run on hardware
//
#include <hip/hip_runtime.h>

// ---------------------------------------------------------------------------
// CDNA5 (gfx1250) implementation of the charge-equilibration NN potential.
// MLP GEMMs use v_wmma_f32_16x16x32_bf16 (wave32, f32 accumulate) with a
// species-sorted row permutation so each 16-row tile uses one expert's
// weights (1x FLOPs instead of the reference's 4x expert overcompute).
// Each wave computes 16 rows x (NT*16) cols, reusing the A fragment across
// NT WMMAs per K-step (A is the dominant traffic stream).
// ---------------------------------------------------------------------------

typedef __attribute__((ext_vector_type(16))) __bf16 v16bf;
typedef __attribute__((ext_vector_type(8)))  __bf16 v8bf;
typedef __attribute__((ext_vector_type(8)))  float  v8f;

#define NM     512
#define NA     64
#define ATOMS  (NM * NA)          // 32768
#define RT     2052               // row tiles (covers 32768 + 4*15 pad, /16)
#define RPAD   (RT * 16)          // 32832 padded rows
#define KF     448                // 418 features padded to multiple of 32
#define AEVL   384
#define N0     256
#define N1     192
#define N2     160

__constant__ float c_sigma[8] = {0.5515909f, 1.8886297f, 1.3225029f, 1.2316629f,
                                 2.1884933f, 1.7750372f, 1.3677907f, 1.3820058f};

__device__ __forceinline__ float celu01(float x) {
  return x > 0.f ? x : 0.1f * (__expf(x * 10.f) - 1.f);
}

// ---------------------------- permutation build ----------------------------

__global__ void k_initperm(int* perm, int* counts) {
  int i = blockIdx.x * blockDim.x + threadIdx.x;
  if (i < RPAD) perm[i] = -1;
  if (i < 4) counts[i] = 0;
}

__global__ void k_count(const int* __restrict__ species, int* counts) {
  int i = blockIdx.x * blockDim.x + threadIdx.x;
  if (i >= ATOMS) return;
  int s = species[i];
  if (s >= 0) atomicAdd(&counts[s], 1);
}

__global__ void k_segments(const int* __restrict__ counts, int* segstart, int* fill,
                           int* tile_species) {
  __shared__ int ss[5];
  if (threadIdx.x == 0) {
    int acc = 0;
    for (int s = 0; s < 4; ++s) {
      ss[s] = acc;
      segstart[s] = acc;
      fill[s] = 0;
      acc += (counts[s] + 15) & ~15;     // 16-row aligned segments
    }
    ss[4] = acc;
  }
  __syncthreads();
  for (int t = threadIdx.x; t < RT; t += blockDim.x) {
    int row = t * 16, sp = 0;
    for (int s = 0; s < 4; ++s)
      if (row >= ss[s] && row < ss[s + 1]) sp = s;
    tile_species[t] = sp;
  }
}

__global__ void k_scatter(const int* __restrict__ species, const int* __restrict__ segstart,
                          int* fill, int* perm) {
  int i = blockIdx.x * blockDim.x + threadIdx.x;
  if (i >= ATOMS) return;
  int s = species[i];
  if (s < 0) return;
  int pos = segstart[s] + atomicAdd(&fill[s], 1);
  perm[pos] = i;
}

// ------------------------------- geometry ----------------------------------

__global__ void k_geom(const float* __restrict__ coords, float* __restrict__ d) {
  int m = blockIdx.x;
  __shared__ float cx[NA], cy[NA], cz[NA];
  int t = threadIdx.x;
  if (t < NA) {
    cx[t] = coords[(m * NA + t) * 3 + 0];
    cy[t] = coords[(m * NA + t) * 3 + 1];
    cz[t] = coords[(m * NA + t) * 3 + 2];
  }
  __syncthreads();
  for (int e = t; e < NA * NA; e += blockDim.x) {
    int i = e >> 6, j = e & 63;
    float dx = cx[i] - cx[j], dy = cy[i] - cy[j], dz = cz[i] - cz[j];
    float dd = sqrtf(dx * dx + dy * dy + dz * dz + 1e-16f) / 0.529177249f; // Bohr
    d[(size_t)m * NA * NA + e] = dd;
  }
}

__global__ void k_atomprep(const int* __restrict__ species, float* sig, float* invj, float* pm) {
  int i = blockIdx.x * blockDim.x + threadIdx.x;
  if (i >= ATOMS) return;
  int s = species[i];
  float sg = c_sigma[s < 0 ? 7 : s];          // -1 wraps to last entry
  sig[i]  = sg;
  invj[i] = 1.7724539f * sg;                  // sqrt(pi) * sigma
  pm[i]   = s >= 0 ? 1.f : 0.f;
}

__global__ void k_zero(float* p, int n) {
  int i = blockIdx.x * blockDim.x + threadIdx.x;
  if (i < n) p[i] = 0.f;
}

// ----------------------- bf16 transposed weight prep -----------------------

// W: [4][Kreal][N] f32 -> Wt: [4][N][Kpad] bf16 (zero-padded K)
__global__ void k_wconv(const float* __restrict__ W, __bf16* __restrict__ Wt,
                        int Kreal, int Kpad, int N) {
  size_t tot = (size_t)4 * N * Kpad;
  for (size_t e = (size_t)blockIdx.x * blockDim.x + threadIdx.x; e < tot;
       e += (size_t)gridDim.x * blockDim.x) {
    int k = (int)(e % Kpad);
    size_t tmp = e / Kpad;
    int n = (int)(tmp % N);
    int s = (int)(tmp / N);
    float v = (k < Kreal) ? W[((size_t)s * Kreal + k) * N + n] : 0.f;
    Wt[e] = (__bf16)v;
  }
}

// ---------------------------- feature assembly -----------------------------

// One block per padded row, 448 threads (one per feature column).
__global__ void k_feats(const float* __restrict__ aev, const float* __restrict__ qraev,
                        const float* __restrict__ q, const float* __restrict__ esp,
                        const int* __restrict__ perm, __bf16* __restrict__ F) {
  int p = blockIdx.x, t = threadIdx.x;
  int r = perm[p];
  float v = 0.f;
  if (r >= 0) {
    if (t < AEVL)            v = aev[(size_t)r * AEVL + t];
    else if (t < AEVL + 32)  v = qraev[(size_t)r * 32 + (t - AEVL)];
    else if (t == AEVL + 32) v = q[r];
    else if (t == AEVL + 33) v = esp[r];
    // t in [418,448): zero pad
  }
  F[(size_t)p * KF + t] = (__bf16)v;
}

// --------------------------- WMMA GEMM + celu ------------------------------

// One wave per 16x(NT*16) output tile; A fragment reused across NT WMMAs.
// X: [RPAD][K] bf16 (species-sorted rows), Wt: [4][N][K] bf16,
// bias: [4][N] f32, Y: [RPAD][N] bf16 (celu applied).
// A frag: lane holds row m, contiguous K run of 16 (16*half offset).
// B frag: lane holds column n, contiguous K run of 16 (16*half offset).
template <int NT>
__global__ __launch_bounds__(32) void k_gemm(const __bf16* __restrict__ X,
                                             const __bf16* __restrict__ Wt,
                                             const float* __restrict__ bias,
                                             const int* __restrict__ tsp,
                                             __bf16* __restrict__ Y,
                                             int K, int N) {
  const int rt = blockIdx.x, ng = blockIdx.y;
  const int lane = threadIdx.x;
  const int half = lane >> 4, idx = lane & 15;
  const int s = tsp[rt];
  const int n0 = ng * (16 * NT) + idx;
  const __bf16* arow = X + (size_t)(rt * 16 + idx) * K + 16 * half;
  const __bf16* bcol[NT];
#pragma unroll
  for (int t = 0; t < NT; ++t)
    bcol[t] = Wt + ((size_t)s * N + n0 + 16 * t) * K + 16 * half;

  v8f acc[NT];
#pragma unroll
  for (int t = 0; t < NT; ++t)
#pragma unroll
    for (int e = 0; e < 8; ++e) acc[t][e] = 0.f;

  for (int kb = 0; kb < K; kb += 32) {
    v16bf a = *(const v16bf*)(arow + kb);
#pragma unroll
    for (int t = 0; t < NT; ++t) {
      v16bf b = *(const v16bf*)(bcol[t] + kb);
      acc[t] = __builtin_amdgcn_wmma_f32_16x16x32_bf16(false, a, false, b,
                                                       (short)0, acc[t],
                                                       false, false);
    }
  }

#pragma unroll
  for (int t = 0; t < NT; ++t) {
    const int n = n0 + 16 * t;
    const float bn = bias[s * N + n];
    __bf16* yb = Y + (size_t)(rt * 16 + 8 * half) * N + n;
#pragma unroll
    for (int e = 0; e < 8; ++e)
      yb[(size_t)e * N] = (__bf16)celu01(acc[t][e] + bn);
  }
}

// Final 160 -> 1 layer as a plain dot product, scattered to original order.
__global__ void k_layer3(const __bf16* __restrict__ H2, const float* __restrict__ W3,
                         const float* __restrict__ b3, const int* __restrict__ perm,
                         const int* __restrict__ tsp, float* __restrict__ out) {
  int p = blockIdx.x * blockDim.x + threadIdx.x;
  if (p >= RPAD) return;
  int r = perm[p];
  if (r < 0) return;
  int s = tsp[p >> 4];
  const __bf16* h = H2 + (size_t)p * N2;
  const float* w = W3 + (size_t)s * N2;
  float acc = b3[s];
  for (int k = 0; k < N2; ++k) acc += (float)h[k] * w[k];
  out[r] = acc;
}

// --------------------------- physics kernels -------------------------------

__global__ void k_qeq(const float* __restrict__ chi, const float* __restrict__ invj,
                      const float* __restrict__ pm, const float* __restrict__ netq,
                      float* __restrict__ q) {
  int m = blockIdx.x, i = threadIdx.x;
  int gi = m * NA + i;
  float c = chi[gi], ij = invj[gi], p = pm[gi];
  __shared__ float s1[NA], s2[NA];
  s1[i] = c * ij;
  s2[i] = ij * p;
  __syncthreads();
  for (int o = 32; o > 0; o >>= 1) {
    if (i < o) { s1[i] += s1[i + o]; s2[i] += s2[i + o]; }
    __syncthreads();
  }
  float corr = (netq[m] + s1[0]) / s2[0];
  q[gi] = -ij * (c - corr) * p;
}

__global__ void k_esp(const float* __restrict__ d, const float* __restrict__ q,
                      const float* __restrict__ sig, const float* __restrict__ pm,
                      float* __restrict__ esp) {
  int m = blockIdx.x, i = threadIdx.x;
  int gi = m * NA + i;
  __shared__ float qs[NA], ss[NA], ps[NA];
  qs[i] = q[gi]; ss[i] = sig[gi]; ps[i] = pm[gi];
  __syncthreads();
  float si = ss[i], acc = 0.f;
  const float* drow = d + (size_t)m * NA * NA + i * NA;
  for (int j = 0; j < NA; ++j) {
    if (j == i) continue;
    float s2 = fmaxf(si * si + ss[j] * ss[j], 1e-8f);
    float dd = drow[j];
    acc += qs[j] * erff(dd * rsqrtf(2.f * s2)) / dd * ps[j];
  }
  esp[gi] = acc * ps[i];
}

__global__ void k_qraev(const float* __restrict__ d, const float* __restrict__ q,
                        const float* __restrict__ pm, float* __restrict__ qraev) {
  int m = blockIdx.x, t = threadIdx.x;               // 256 threads
  int i = t >> 2, r0 = (t & 3) * 8;
  __shared__ float qs[NA], ps[NA];
  if (t < NA) { qs[t] = q[m * NA + t]; ps[t] = pm[m * NA + t]; }
  __syncthreads();
  float pi_ = ps[i];
  float acc[8] = {0.f, 0.f, 0.f, 0.f, 0.f, 0.f, 0.f, 0.f};
  const float* drow = d + (size_t)m * NA * NA + i * NA;
  for (int j = 0; j < NA; ++j) {
    if (j == i) continue;
    float dd = drow[j];
    float fc = dd < 10.f ? 0.5f * __cosf(0.31415926535f * dd) + 0.5f : 0.f;
    float w = 0.25f * fc * pi_ * ps[j] * qs[j];
    if (w != 0.f) {
#pragma unroll
      for (int rr = 0; rr < 8; ++rr) {
        float rs = 0.8f + (float)(r0 + rr) * (8.2f / 31.f);
        float dmr = dd - rs;
        acc[rr] += w * __expf(-4.f * dmr * dmr);
      }
    }
  }
  float* out = qraev + (size_t)(m * NA + i) * 32 + r0;
#pragma unroll
  for (int rr = 0; rr < 8; ++rr) out[rr] = acc[rr];
}

__global__ void k_energy(const float* __restrict__ ae, const float* __restrict__ q,
                         const float* __restrict__ d, float* __restrict__ energy) {
  int m = blockIdx.x, i = threadIdx.x;
  int gi = m * NA + i;
  __shared__ float qs[NA], red[NA];
  qs[i] = q[gi];
  __syncthreads();
  float acc = ae[gi];
  const float* drow = d + (size_t)m * NA * NA + i * NA;
  float qi = qs[i];
  for (int j = i + 1; j < NA; ++j) {
    float dd = drow[j];
    float sc = 1.f / (1.f + __expf(-(dd - 2.2f) * 8.5f));   // elect_screen
    acc += qi * qs[j] * sc / dd;                            // q=0 masks padding
  }
  red[i] = acc;
  __syncthreads();
  for (int o = 32; o > 0; o >>= 1) {
    if (i < o) red[i] += red[i + o];
    __syncthreads();
  }
  if (i == 0) energy[m] = red[0];
}

// (species bits, energy, q) packed flat into d_out.
__global__ void k_output(const int* __restrict__ species, const float* __restrict__ energy,
                         const float* __restrict__ q, float* __restrict__ out) {
  int i = blockIdx.x * blockDim.x + threadIdx.x;
  if (i < ATOMS) {
    reinterpret_cast<int*>(out)[i] = species[i];
  } else if (i < ATOMS + NM) {
    out[i] = energy[i - ATOMS];
  } else if (i < ATOMS + NM + ATOMS) {
    out[i] = q[i - ATOMS - NM];
  }
}

// ------------------------------- host side ---------------------------------

extern "C" void kernel_launch(void* const* d_in, const int* in_sizes, int n_in,
                              void* d_out, int out_size, void* d_ws, size_t ws_size,
                              hipStream_t stream) {
  (void)in_sizes; (void)n_in; (void)out_size; (void)ws_size;
  const int*   species = (const int*)d_in[0];
  const float* coords  = (const float*)d_in[1];
  const float* netq    = (const float*)d_in[2];
  const float* aev     = (const float*)d_in[3];
  const float* chiW0 = (const float*)d_in[4],  *chiB0 = (const float*)d_in[5];
  const float* chiW1 = (const float*)d_in[6],  *chiB1 = (const float*)d_in[7];
  const float* chiW2 = (const float*)d_in[8],  *chiB2 = (const float*)d_in[9];
  const float* chiW3 = (const float*)d_in[10], *chiB3 = (const float*)d_in[11];
  const float* aniW0 = (const float*)d_in[12], *aniB0 = (const float*)d_in[13];
  const float* aniW1 = (const float*)d_in[14], *aniB1 = (const float*)d_in[15];
  const float* aniW2 = (const float*)d_in[16], *aniB2 = (const float*)d_in[17];
  const float* aniW3 = (const float*)d_in[18], *aniB3 = (const float*)d_in[19];

  char* cur = (char*)d_ws;
  auto alloc = [&](size_t bytes) -> void* {
    void* p = (void*)cur;
    cur += (bytes + 255) & ~(size_t)255;
    return p;
  };
  float*  dD     = (float*)alloc((size_t)NM * NA * NA * 4);
  float*  dSig   = (float*)alloc((size_t)ATOMS * 4);
  float*  dInvj  = (float*)alloc((size_t)ATOMS * 4);
  float*  dPm    = (float*)alloc((size_t)ATOMS * 4);
  float*  dQ     = (float*)alloc((size_t)ATOMS * 4);
  float*  dEsp   = (float*)alloc((size_t)ATOMS * 4);
  float*  dChi   = (float*)alloc((size_t)ATOMS * 4);   // reused as ae
  float*  dQraev = (float*)alloc((size_t)ATOMS * 32 * 4);
  int*    dCounts = (int*)alloc(4 * 4);
  int*    dSeg    = (int*)alloc(4 * 4);
  int*    dFill   = (int*)alloc(4 * 4);
  int*    dPerm   = (int*)alloc((size_t)RPAD * 4);
  int*    dTsp    = (int*)alloc((size_t)RT * 4);
  __bf16* dF      = (__bf16*)alloc((size_t)RPAD * KF * 2);
  __bf16* dH0     = (__bf16*)alloc((size_t)RPAD * N0 * 2);
  __bf16* dH1     = (__bf16*)alloc((size_t)RPAD * N1 * 2);
  __bf16* dH2     = (__bf16*)alloc((size_t)RPAD * N2 * 2);
  __bf16* dWtC0   = (__bf16*)alloc((size_t)4 * N0 * KF * 2);
  __bf16* dWtC1   = (__bf16*)alloc((size_t)4 * N1 * N0 * 2);
  __bf16* dWtC2   = (__bf16*)alloc((size_t)4 * N2 * N1 * 2);
  __bf16* dWtA0   = (__bf16*)alloc((size_t)4 * N0 * KF * 2);
  __bf16* dWtA1   = (__bf16*)alloc((size_t)4 * N1 * N0 * 2);
  __bf16* dWtA2   = (__bf16*)alloc((size_t)4 * N2 * N1 * 2);
  float*  dEnergy = (float*)alloc((size_t)NM * 4);

  // permutation (species-sorted, 16-aligned segments)
  k_initperm<<<(RPAD + 255) / 256, 256, 0, stream>>>(dPerm, dCounts);
  k_count<<<(ATOMS + 255) / 256, 256, 0, stream>>>(species, dCounts);
  k_segments<<<1, 256, 0, stream>>>(dCounts, dSeg, dFill, dTsp);
  k_scatter<<<(ATOMS + 255) / 256, 256, 0, stream>>>(species, dSeg, dFill, dPerm);

  // geometry + per-atom constants, zero charge state
  k_geom<<<NM, 256, 0, stream>>>(coords, dD);
  k_atomprep<<<(ATOMS + 255) / 256, 256, 0, stream>>>(species, dSig, dInvj, dPm);
  k_zero<<<(ATOMS + 255) / 256, 256, 0, stream>>>(dQ, ATOMS);
  k_zero<<<(ATOMS + 255) / 256, 256, 0, stream>>>(dEsp, ATOMS);
  k_zero<<<(ATOMS * 32 + 255) / 256, 256, 0, stream>>>(dQraev, ATOMS * 32);

  // bf16 transposed weights (L2-resident)
  k_wconv<<<512, 256, 0, stream>>>(chiW0, dWtC0, 418, KF, N0);
  k_wconv<<<512, 256, 0, stream>>>(chiW1, dWtC1, N0, N0, N1);
  k_wconv<<<512, 256, 0, stream>>>(chiW2, dWtC2, N1, N1, N2);
  k_wconv<<<512, 256, 0, stream>>>(aniW0, dWtA0, 418, KF, N0);
  k_wconv<<<512, 256, 0, stream>>>(aniW1, dWtA1, N0, N0, N1);
  k_wconv<<<512, 256, 0, stream>>>(aniW2, dWtA2, N1, N1, N2);

  auto mlp_pass = [&](const __bf16* w0, const __bf16* w1, const __bf16* w2,
                      const float* b0, const float* b1, const float* b2,
                      const float* w3, const float* b3, float* outv) {
    k_feats<<<RPAD, KF, 0, stream>>>(aev, dQraev, dQ, dEsp, dPerm, dF);
    k_gemm<4><<<dim3(RT, N0 / 64), 32, 0, stream>>>(dF,  w0, b0, dTsp, dH0, KF, N0);
    k_gemm<4><<<dim3(RT, N1 / 64), 32, 0, stream>>>(dH0, w1, b1, dTsp, dH1, N0, N1);
    k_gemm<2><<<dim3(RT, N2 / 32), 32, 0, stream>>>(dH1, w2, b2, dTsp, dH2, N1, N2);
    k_zero<<<(ATOMS + 255) / 256, 256, 0, stream>>>(outv, ATOMS);
    k_layer3<<<(RPAD + 127) / 128, 128, 0, stream>>>(dH2, w3, b3, dPerm, dTsp, outv);
  };

  // charge-equilibration loop (2 iterations)
  for (int it = 0; it < 2; ++it) {
    mlp_pass(dWtC0, dWtC1, dWtC2, chiB0, chiB1, chiB2, chiW3, chiB3, dChi);
    k_qeq<<<NM, NA, 0, stream>>>(dChi, dInvj, dPm, netq, dQ);
    k_esp<<<NM, NA, 0, stream>>>(dD, dQ, dSig, dPm, dEsp);
    k_qraev<<<NM, 256, 0, stream>>>(dD, dQ, dPm, dQraev);
  }

  // final atomic-energy pass + screened Coulomb
  mlp_pass(dWtA0, dWtA1, dWtA2, aniB0, aniB1, aniB2, aniW3, aniB3, dChi);
  k_energy<<<NM, NA, 0, stream>>>(dChi, dQ, dD, dEnergy);

  k_output<<<(2 * ATOMS + NM + 255) / 256, 256, 0, stream>>>(species, dEnergy, dQ,
                                                             (float*)d_out);
}
